// MultiHeadAttention_49770081026139
// MI455X (gfx1250) — matrix operand order repaired and checked by reading, not compiled, not run on hardware
//
#include <hip/hip_runtime.h>

// ---------------------------------------------------------------------------
// Problem constants (reference: b=4, n=256, H=512, heads=16, d=32)
// ---------------------------------------------------------------------------
#define B_   4
#define N_   256
#define H_   512
#define NH_  16
#define DH_  32
#define MT_  64          // rows per block for the big GEMMs
#define SA_  264         // LDS half-stride for a 64x256 staged K-half (256+8 pad)

typedef __attribute__((ext_vector_type(16))) _Float16 v16h;
typedef __attribute__((ext_vector_type(8)))  _Float16 v8h;
typedef __attribute__((ext_vector_type(4)))  _Float16 v4h;
typedef __attribute__((ext_vector_type(8)))  float    v8f;
typedef __attribute__((ext_vector_type(4)))  float    v4f;

__device__ __forceinline__ v8f wmma16(v16h a, v16h b, v8f c) {
    return __builtin_amdgcn_wmma_f32_16x16x32_f16(false, a, false, b,
                                                  (short)0, c, false, false);
}

__device__ __forceinline__ v16h cat8(v8h lo, v8h hi) {
    return __builtin_shufflevector(lo, hi, 0,1,2,3,4,5,6,7,8,9,10,11,12,13,14,15);
}

// A fragment (16x32 f16) from LDS tile (row-major, half-stride SA_), K-offset k0
// ISA layout: lane L -> row L%16, halves [k0+ks .. +7] and [k0+ks+16 .. +23], ks=(L/16)*8
__device__ __forceinline__ v16h a_frag_lds(const _Float16* lds, int mt, int lane, int k0) {
    const int r = mt * 16 + (lane & 15);
    const _Float16* p = lds + r * SA_ + k0 + ((lane >> 4) << 3);
    return cat8(*(const v8h*)p, *(const v8h*)(p + 16));
}

// A fragment from global f16 buffer with row stride 1024 halves (qk scratch in wout bytes)
__device__ __forceinline__ v16h a_frag_qk(const _Float16* __restrict__ g, size_t row,
                                          int gk, int lane) {
    const _Float16* p = g + row * 1024 + gk + ((lane >> 4) << 3);
    return cat8(*(const v8h*)p, *(const v8h*)(p + 16));
}

// B fragment (32x16 f16): lane L -> col n, halves [gk+ks .. +7] and [gk+ks+16 .. +23], ks=(L/16)*8
// BT is the transposed f16 weight: BT[n*512 + k] = W[k*512 + n]
__device__ __forceinline__ v16h b_frag(const _Float16* __restrict__ BT, int n,
                                       int gk, int lane) {
    const _Float16* p = BT + (size_t)n * H_ + gk + ((lane >> 4) << 3);
    return cat8(*(const v8h*)p, *(const v8h*)(p + 16));
}

// Stage one K-half (64 rows x 256 cols) f32 -> f16 LDS. src row stride = 512 floats.
__device__ __forceinline__ void stage_half(const float* __restrict__ src,
                                           _Float16* lds, int tid) {
    #pragma unroll
    for (int it = 0; it < 16; ++it) {
        const int idx = it * 256 + tid;   // 0..4095
        const int r   = idx >> 6;         // 0..63
        const int c4  = idx & 63;         // *4 cols
        v4f x = *(const v4f*)(src + (size_t)r * H_ + c4 * 4);
        v4h h;
        h[0] = (_Float16)x[0]; h[1] = (_Float16)x[1];
        h[2] = (_Float16)x[2]; h[3] = (_Float16)x[3];
        *(v4h*)(lds + r * SA_ + c4 * 4) = h;
    }
}

// 8 K-steps (one staged K-half) of the 64x512 GEMM; each wave: 4 mtiles x 4 ntiles
__device__ __forceinline__ void gemm_khalf(const _Float16* lds,
                                           const _Float16* __restrict__ BT,
                                           int lane, int cb, int khbase,
                                           v8f acc[4][4]) {
    #pragma unroll 1
    for (int kk = 0; kk < 8; ++kk) {
        const int k0 = kk * 32;
        const int gk = khbase + k0;
        v16h a0 = a_frag_lds(lds, 0, lane, k0);
        v16h a1 = a_frag_lds(lds, 1, lane, k0);
        v16h a2 = a_frag_lds(lds, 2, lane, k0);
        v16h a3 = a_frag_lds(lds, 3, lane, k0);
        #pragma unroll
        for (int nt = 0; nt < 4; ++nt) {
            v16h b = b_frag(BT, cb + nt * 16 + (lane & 15), gk, lane);
            acc[0][nt] = wmma16(a0, b, acc[0][nt]);
            acc[1][nt] = wmma16(a1, b, acc[1][nt]);
            acc[2][nt] = wmma16(a2, b, acc[2][nt]);
            acc[3][nt] = wmma16(a3, b, acc[3][nt]);
        }
    }
}

// ---------------------------------------------------------------------------
// Weight convert+transpose: dst[w][n*512+k] = (f16) src[w][k*512+n]
// ---------------------------------------------------------------------------
struct WPtrs { const float* src[6]; };

__global__ __launch_bounds__(256) void wconvert(WPtrs wp, _Float16* __restrict__ dst) {
    const float* s = wp.src[blockIdx.y];
    _Float16* d = dst + (size_t)blockIdx.y * H_ * H_;
    const int n = blockIdx.x;
    for (int k = threadIdx.x; k < H_; k += 256)
        d[(size_t)n * H_ + k] = (_Float16)s[(size_t)k * H_ + n];
}

// ---------------------------------------------------------------------------
// Generic 512x512 projection GEMM: C = (A@W + bias) * alpha    (M = 64*gridDim)
// ---------------------------------------------------------------------------
__global__ __launch_bounds__(256) void gemm512(const float* __restrict__ A,
                                               const _Float16* __restrict__ BT,
                                               const float* __restrict__ bias,
                                               float* __restrict__ C, float alpha) {
    __shared__ __align__(16) _Float16 alds[MT_ * SA_];
    const int tid  = threadIdx.x;
    const size_t m0 = (size_t)blockIdx.x * MT_;
    const int lane = tid & 31, wid = tid >> 5;
    const int cb   = wid * 64;
    v8f acc[4][4] = {};
    for (int kh = 0; kh < 2; ++kh) {
        __syncthreads();
        stage_half(A + m0 * H_ + kh * 256, alds, tid);
        __syncthreads();
        gemm_khalf(alds, BT, lane, cb, kh * 256, acc);
    }
    #pragma unroll
    for (int nt = 0; nt < 4; ++nt) {
        const int c = cb + nt * 16 + (lane & 15);
        const float bv = bias[c];
        #pragma unroll
        for (int mt = 0; mt < 4; ++mt) {
            #pragma unroll
            for (int p = 0; p < 8; ++p) {
                const int r = mt * 16 + p + ((lane >> 4) << 3);
                C[(m0 + r) * H_ + c] = (acc[mt][nt][p] + bv) * alpha;
            }
        }
    }
}

// ---------------------------------------------------------------------------
// Fused edge pipeline per block (64 edge rows = one (b, i, j0..j0+63)):
//   eh = e@We+be  ->  qk = eh*q[i,c]*k[j,c]  -> logits (per-head sums, +bias)
//   w_out = qk@Weo + beo      (qk staged as f16 in the w_out rows' own bytes)
// ---------------------------------------------------------------------------
__global__ __launch_bounds__(256) void fused_edge(
        const float* __restrict__ e, const float* __restrict__ qproj,
        const float* __restrict__ kproj, const float* __restrict__ attn_bias,
        const _Float16* __restrict__ WeT, const float* __restrict__ be_,
        const _Float16* __restrict__ WeoT, const float* __restrict__ beo_,
        float* __restrict__ wlog, float* __restrict__ wout) {
    __shared__ __align__(16) _Float16 alds[MT_ * SA_];
    __shared__ float wt[MT_ * NH_];

    const int tid = threadIdx.x;
    const int blk = blockIdx.x;            // b(4) * i(256) * jt(4)
    const int jt  = blk & 3;
    const int i   = (blk >> 2) & 255;
    const int bb  = blk >> 10;
    const int j0  = jt * MT_;
    const size_t mrow0 = ((size_t)bb * N_ + i) * N_ + j0;
    const float* eb = e + mrow0 * H_;
    _Float16* qk = (_Float16*)wout;        // f16 scratch aliased in wout bytes

    wt[tid] = 0.f; wt[tid + 256] = 0.f; wt[tid + 512] = 0.f; wt[tid + 768] = 0.f;

    const int lane = tid & 31, wid = tid >> 5;
    const int cb = wid * 64;

    // ---- phase 1: eh = e @ We ----
    v8f acc[4][4] = {};
    for (int kh = 0; kh < 2; ++kh) {
        __syncthreads();
        stage_half(eb + kh * 256, alds, tid);
        __syncthreads();
        gemm_khalf(alds, WeT, lane, cb, kh * 256, acc);
    }

    // ---- elementwise + per-head logit reduction + qk f16 spill ----
    const float* qrow = qproj + ((size_t)bb * N_ + i) * H_;
    #pragma unroll
    for (int nt = 0; nt < 4; ++nt) {
        const int c = cb + nt * 16 + (lane & 15);
        const float bev = be_[c];
        const float qv  = qrow[c];          // scale already folded into qproj
        const int hh = c >> 5;
        #pragma unroll
        for (int mt = 0; mt < 4; ++mt) {
            #pragma unroll
            for (int p = 0; p < 8; ++p) {
                const int r = mt * 16 + p + ((lane >> 4) << 3);
                const float kv = kproj[((size_t)bb * N_ + j0 + r) * H_ + c];
                const float q_ = (acc[mt][nt][p] + bev) * qv * kv;
                atomicAdd(&wt[r * NH_ + hh], q_);             // ds_add_f32
                qk[(mrow0 + r) * 1024 + c] = (_Float16)q_;
            }
        }
    }
    __threadfence();
    __syncthreads();

    // ---- attention logits + bias ----
    #pragma unroll
    for (int u = 0; u < 4; ++u) {
        const int idx = u * 256 + tid;      // 0..1023
        const int r = idx >> 4, hh = idx & 15;
        const size_t o = (((size_t)bb * NH_ + hh) * N_ + i) * N_ + j0 + r;
        wlog[o] = wt[r * NH_ + hh] + attn_bias[o];
    }

    // ---- phase 2: w_out = qk @ Weo + beo ----
    v8f acc2[4][4] = {};
    #pragma unroll 1
    for (int kk = 0; kk < 16; ++kk) {
        const int gk = kk * 32;
        v16h a0 = a_frag_qk(qk, mrow0 +  0 + (lane & 15), gk, lane);
        v16h a1 = a_frag_qk(qk, mrow0 + 16 + (lane & 15), gk, lane);
        v16h a2 = a_frag_qk(qk, mrow0 + 32 + (lane & 15), gk, lane);
        v16h a3 = a_frag_qk(qk, mrow0 + 48 + (lane & 15), gk, lane);
        #pragma unroll
        for (int nt = 0; nt < 4; ++nt) {
            v16h b = b_frag(WeoT, cb + nt * 16 + (lane & 15), gk, lane);
            acc2[0][nt] = wmma16(a0, b, acc2[0][nt]);
            acc2[1][nt] = wmma16(a1, b, acc2[1][nt]);
            acc2[2][nt] = wmma16(a2, b, acc2[2][nt]);
            acc2[3][nt] = wmma16(a3, b, acc2[3][nt]);
        }
    }
    __syncthreads();   // all qk reads done before overwriting the same bytes
    #pragma unroll
    for (int nt = 0; nt < 4; ++nt) {
        const int c = cb + nt * 16 + (lane & 15);
        const float bv = beo_[c];
        #pragma unroll
        for (int mt = 0; mt < 4; ++mt) {
            #pragma unroll
            for (int p = 0; p < 8; ++p) {
                const int r = mt * 16 + p + ((lane >> 4) << 3);
                wout[(mrow0 + r) * H_ + c] = acc2[mt][nt][p] + bv;
            }
        }
    }
}

// ---------------------------------------------------------------------------
// Softmax over j + x[b,h,i,:] = a @ vh  (one block per (b,h,i))
// ---------------------------------------------------------------------------
__global__ __launch_bounds__(256) void attn_soft(const float* __restrict__ wlog,
                                                 const float* __restrict__ vproj,
                                                 float* __restrict__ xbuf) {
    __shared__ float aa[N_];
    __shared__ float red[256];
    const int tid = threadIdx.x;
    const int blk = blockIdx.x;
    const int i  = blk & 255;
    const int hh = (blk >> 8) & 15;
    const int bb = blk >> 12;
    const float* wrow = wlog + (((size_t)bb * NH_ + hh) * N_ + i) * N_;
    const float v = wrow[tid];
    red[tid] = v;
    __syncthreads();
    #pragma unroll
    for (int s = 128; s > 0; s >>= 1) {
        if (tid < s) red[tid] = fmaxf(red[tid], red[tid + s]);
        __syncthreads();
    }
    const float mx = red[0];
    __syncthreads();
    const float ex = __expf(v - mx);
    aa[tid] = ex;
    red[tid] = ex;
    __syncthreads();
    #pragma unroll
    for (int s = 128; s > 0; s >>= 1) {
        if (tid < s) red[tid] += red[tid + s];
        __syncthreads();
    }
    const float inv = 1.0f / red[0];
    __syncthreads();
    const int dd = tid & 31, seg = tid >> 5;
    float p = 0.f;
    const float* vbase = vproj + (size_t)bb * N_ * H_ + hh * DH_ + dd;
    #pragma unroll 1
    for (int j = seg * 32; j < seg * 32 + 32; ++j)
        p += aa[j] * vbase[(size_t)j * H_];
    red[tid] = p;
    __syncthreads();
    if (tid < 32) {
        float s = 0.f;
        #pragma unroll
        for (int g = 0; g < 8; ++g) s += red[g * 32 + tid];
        xbuf[((size_t)bb * N_ + i) * H_ + hh * DH_ + tid] = s * inv;
    }
}

// ---------------------------------------------------------------------------
extern "C" void kernel_launch(void* const* d_in, const int* in_sizes, int n_in,
                              void* d_out, int out_size, void* d_ws, size_t ws_size,
                              hipStream_t stream) {
    (void)in_sizes; (void)n_in; (void)out_size; (void)ws_size;
    const float* q   = (const float*)d_in[0];
    const float* k   = (const float*)d_in[1];
    const float* v   = (const float*)d_in[2];
    const float* e   = (const float*)d_in[3];
    const float* ab  = (const float*)d_in[4];
    const float* Wq  = (const float*)d_in[6];
    const float* bq  = (const float*)d_in[7];
    const float* Wk  = (const float*)d_in[8];
    const float* bk  = (const float*)d_in[9];
    const float* Wv  = (const float*)d_in[10];
    const float* bv  = (const float*)d_in[11];
    const float* We  = (const float*)d_in[12];
    const float* be  = (const float*)d_in[13];
    const float* Wo  = (const float*)d_in[14];
    const float* bo  = (const float*)d_in[15];
    const float* Weo = (const float*)d_in[16];
    const float* beo = (const float*)d_in[17];

    float* out  = (float*)d_out;
    float* xout = out;                                   // (b, n, H)
    float* wout = out + (size_t)B_ * N_ * H_;            // (b, n, n, H)

    char* ws = (char*)d_ws;
    _Float16* wts = (_Float16*)ws;                       // 6 transposed f16 weights
    const size_t WSZ = (size_t)H_ * H_;                  // halves per weight
    float* qproj = (float*)(ws + 6 * WSZ * 2);
    float* kproj = qproj + (size_t)B_ * N_ * H_;
    float* vproj = kproj + (size_t)B_ * N_ * H_;
    float* xbuf  = vproj + (size_t)B_ * N_ * H_;
    float* wlog  = xbuf  + (size_t)B_ * N_ * H_;         // (b, h, n, n) 16.8 MB

    WPtrs wp;
    wp.src[0] = Wq; wp.src[1] = Wk; wp.src[2] = Wv;
    wp.src[3] = We; wp.src[4] = Wo; wp.src[5] = Weo;
    wconvert<<<dim3(H_, 6, 1), 256, 0, stream>>>(wp, wts);

    const float scale = 0.17677669529663688f;            // 1/sqrt(32)
    gemm512<<<(B_ * N_) / MT_, 256, 0, stream>>>(q, wts + 0 * WSZ, bq, qproj, scale);
    gemm512<<<(B_ * N_) / MT_, 256, 0, stream>>>(k, wts + 1 * WSZ, bk, kproj, 1.f);
    gemm512<<<(B_ * N_) / MT_, 256, 0, stream>>>(v, wts + 2 * WSZ, bv, vproj, 1.f);

    fused_edge<<<B_ * N_ * (N_ / MT_), 256, 0, stream>>>(
        e, qproj, kproj, ab, wts + 3 * WSZ, be, wts + 5 * WSZ, beo, wlog, wout);

    attn_soft<<<B_ * NH_ * N_, 256, 0, stream>>>(wlog, vproj, xbuf);

    gemm512<<<(B_ * N_) / MT_, 256, 0, stream>>>(xbuf, wts + 4 * WSZ, bo, xout, 1.f);
}
